// StyleLoss_68616397521363
// MI455X (gfx1250) — matrix operand order repaired and checked by reading, not compiled, not run on hardware
//
#include <hip/hip_runtime.h>
#include <hip/hip_bf16.h>

// ---------------- problem constants ----------------
#define CCH   256            // channels
#define HH    96
#define WW    96
#define KP    3              // patch size
#define HO    94             // H - 2
#define WO    94
#define DDIM  2304           // C*3*3
#define LL    8836           // 94*94 patches
#define LP    8960           // padded: multiple of 256 (N tile) and 128 (M tile)
#define EPSN  1e-8f

// workspace layout (bytes, all 256-aligned)
#define AB_BYTES   ((size_t)LP * DDIM * 2)            // 41,287,680 each
#define BEST_OFF   (2 * AB_BYTES)                     // u64[LP]
#define BEST_BYTES ((size_t)LP * 8)
#define PART_OFF   (BEST_OFF + BEST_BYTES)            // float[LL]

typedef __attribute__((ext_vector_type(16))) __bf16 v16bf;
typedef __attribute__((ext_vector_type(8)))  float  v8f;

union ABFrag { uint4 q[2]; v16bf v; };

static __device__ __forceinline__ unsigned short f32_to_bf16_rne(float f) {
    unsigned int u = __float_as_uint(f);
    unsigned int r = u + 0x7FFFu + ((u >> 16) & 1u);   // round-to-nearest-even
    return (unsigned short)(r >> 16);
}

static __device__ __forceinline__ unsigned int f32_sortable(float f) {
    unsigned int u = __float_as_uint(f);
    return (u & 0x80000000u) ? ~u : (u | 0x80000000u); // monotonic float->u32
}

// ---------------------------------------------------------------------------
// Kernel 1: per-patch L2 norm + bf16 normalized patch rows.
// grid.x = LP (padded rows zeroed), grid.y = 0 -> x/A, 1 -> style/B
// Row p of dst = normalized patch p, D contiguous bf16 (patch-major).
// ---------------------------------------------------------------------------
__global__ __launch_bounds__(256)
void build_patches(const float* __restrict__ x, const float* __restrict__ style,
                   unsigned short* __restrict__ Abf, unsigned short* __restrict__ Bbf)
{
    const int p = blockIdx.x;
    const float* src = blockIdx.y ? style : x;
    unsigned short* dst = blockIdx.y ? Bbf : Abf;
    const int t = threadIdx.x;
    __shared__ float red[256];

    if (p >= LL) {                       // zero the padding rows
        for (int d = t; d < DDIM; d += 256) dst[(size_t)p * DDIM + d] = 0;
        return;
    }
    const int ho = p / WO, wo = p % WO;

    float vals[9];                       // DDIM/256 == 9 exactly
    float acc = 0.f;
    int n = 0;
    for (int d = t; d < DDIM; d += 256, ++n) {
        const int c = d / 9, r = d % 9, kh = r / 3, kw = r % 3;
        const float v = src[(size_t)c * (HH * WW) + (ho + kh) * WW + (wo + kw)];
        vals[n] = v;
        acc += v * v;
    }
    red[t] = acc; __syncthreads();
    for (int s = 128; s > 0; s >>= 1) { if (t < s) red[t] += red[t + s]; __syncthreads(); }
    const float inv = 1.0f / (sqrtf(red[0]) + EPSN);

    n = 0;
    for (int d = t; d < DDIM; d += 256, ++n)
        dst[(size_t)p * DDIM + d] = f32_to_bf16_rne(vals[n] * inv);
}

// ---------------------------------------------------------------------------
// Kernel 2: init packed argmax accumulators
// ---------------------------------------------------------------------------
__global__ void init_best(unsigned long long* __restrict__ best)
{
    const int i = blockIdx.x * blockDim.x + threadIdx.x;
    if (i < LP) best[i] = 0ull;
}

// ---------------------------------------------------------------------------
// Kernel 3: bf16 WMMA GEMM (ipn^T x spn) with fused per-row argmax.
// Block = 256 threads = 8 waves in 2(M) x 4(N); wave tile 64x64:
//   16 accumulator fragments (128 VGPRs), 4 A + 4 B fragments per K step,
//   16 WMMAs per 16 global_load_b128 -> 512 B of L2 traffic per WMMA.
// Workgroup tile 128(M) x 256(N). Grid = (LP/256, LP/128) = (35, 70).
// K = DDIM = 72 steps of 32.
// ---------------------------------------------------------------------------
__global__ __launch_bounds__(256)
void gemm_argmax(const unsigned short* __restrict__ Abf,
                 const unsigned short* __restrict__ Bbf,
                 unsigned long long* __restrict__ best)
{
    const int lane  = threadIdx.x & 31;
    const int w     = threadIdx.x >> 5;
    const int waveM = w & 1;      // 2 waves along M
    const int waveN = w >> 1;     // 4 waves along N
    const int i0 = blockIdx.y * 128 + waveM * 64;
    const int j0 = blockIdx.x * 256 + waveN * 64;
    const int lr = lane & 15;     // row (A) or col (B/C/D) within 16
    const int lh = lane >> 4;     // K-half selector

    v8f c[4][4] = {};

    for (int k0 = 0; k0 < DDIM; k0 += 32) {
        ABFrag a[4], b[4];
        // A 16x32 bf16: lane holds row M=lr, K = lh*16 .. lh*16+15 contiguous.
        #pragma unroll
        for (int mi = 0; mi < 4; ++mi) {
            const unsigned short* p = Abf + (size_t)(i0 + mi * 16 + lr) * DDIM + k0 + lh * 16;
            a[mi].q[0] = *reinterpret_cast<const uint4*>(p);
            a[mi].q[1] = *reinterpret_cast<const uint4*>(p + 8);
        }
        // B 32x16 bf16: lane holds col N=lr, K = lh*16 .. lh*16+15 contiguous.
        #pragma unroll
        for (int ni = 0; ni < 4; ++ni) {
            const unsigned short* p = Bbf + (size_t)(j0 + ni * 16 + lr) * DDIM + k0 + lh * 16;
            b[ni].q[0] = *reinterpret_cast<const uint4*>(p);
            b[ni].q[1] = *reinterpret_cast<const uint4*>(p + 8);
        }
        #pragma unroll
        for (int mi = 0; mi < 4; ++mi)
            #pragma unroll
            for (int ni = 0; ni < 4; ++ni)
                c[mi][ni] = __builtin_amdgcn_wmma_f32_16x16x32_bf16(
                    false, a[mi].v, false, b[ni].v,
                    (short)0, c[mi][ni], false, false);
    }

    // Fused argmax epilogue.
    // C layout: VGPR r, lanes 0-15 -> (M = r, N = lr); lanes 16-31 -> (M = r+8, N = lr).
    #pragma unroll
    for (int mi = 0; mi < 4; ++mi) {
        #pragma unroll
        for (int r = 0; r < 8; ++r) {
            unsigned long long pk = 0ull;
            #pragma unroll
            for (int ni = 0; ni < 4; ++ni) {
                const int j = j0 + ni * 16 + lr;
                if (j < LL) {
                    const float v = c[mi][ni][r];
                    const unsigned long long q =
                        ((unsigned long long)f32_sortable(v) << 32) | (unsigned int)j;
                    if (q > pk) pk = q;
                }
            }
            // butterfly over the 16 lanes of each row half (stays within half)
            #pragma unroll
            for (int off = 1; off < 16; off <<= 1) {
                const unsigned long long q = __shfl_xor(pk, off, 32);
                if (q > pk) pk = q;
            }
            if (lr == 0) {
                const int i = i0 + mi * 16 + r + lh * 8;
                if (i < LL) atomicMax(best + i, pk);
            }
        }
    }
}

// ---------------------------------------------------------------------------
// Kernel 4: gather matched style patch, per-patch squared error partial sum.
// One block per input patch; reads original f32 data directly (no unfold).
// ---------------------------------------------------------------------------
__global__ __launch_bounds__(256)
void gather_mse(const float* __restrict__ x, const float* __restrict__ style,
                const unsigned long long* __restrict__ best,
                float* __restrict__ partials)
{
    const int i = blockIdx.x;
    const int t = threadIdx.x;
    __shared__ float red[256];

    unsigned int j = (unsigned int)(best[i] & 0xFFFFFFFFu);
    if (j >= LL) j = 0;
    const int ho = i / WO, wo = i % WO;
    const int h2 = (int)j / WO, w2 = (int)j % WO;

    float acc = 0.f;
    for (int d = t; d < DDIM; d += 256) {
        const int c = d / 9, r = d % 9, kh = r / 3, kw = r % 3;
        const float a = x[(size_t)c * (HH * WW) + (ho + kh) * WW + (wo + kw)];
        const float b = style[(size_t)c * (HH * WW) + (h2 + kh) * WW + (w2 + kw)];
        const float df = a - b;
        acc += df * df;
    }
    red[t] = acc; __syncthreads();
    for (int s = 128; s > 0; s >>= 1) { if (t < s) red[t] += red[t + s]; __syncthreads(); }
    if (t == 0) partials[i] = red[0];
}

// ---------------------------------------------------------------------------
// Kernel 5: deterministic single-block final reduction + mean.
// ---------------------------------------------------------------------------
__global__ __launch_bounds__(256)
void finalize(const float* __restrict__ partials, float* __restrict__ out)
{
    __shared__ double red[256];
    const int t = threadIdx.x;
    double acc = 0.0;
    for (int i = t; i < LL; i += 256) acc += (double)partials[i];
    red[t] = acc; __syncthreads();
    for (int s = 128; s > 0; s >>= 1) { if (t < s) red[t] += red[t + s]; __syncthreads(); }
    if (t == 0) out[0] = (float)(red[0] / ((double)DDIM * (double)LL));
}

// ---------------------------------------------------------------------------
extern "C" void kernel_launch(void* const* d_in, const int* in_sizes, int n_in,
                              void* d_out, int out_size, void* d_ws, size_t ws_size,
                              hipStream_t stream)
{
    const float* x     = (const float*)d_in[0];
    const float* style = (const float*)d_in[1];
    float* out = (float*)d_out;

    char* ws = (char*)d_ws;
    unsigned short*     Abf      = (unsigned short*)ws;
    unsigned short*     Bbf      = (unsigned short*)(ws + AB_BYTES);
    unsigned long long* best     = (unsigned long long*)(ws + BEST_OFF);
    float*              partials = (float*)(ws + PART_OFF);

    // 1) normalized bf16 patch matrices (both tensors in one launch)
    build_patches<<<dim3(LP, 2), 256, 0, stream>>>(x, style, Abf, Bbf);
    // 2) init argmax accumulators
    init_best<<<(LP + 255) / 256, 256, 0, stream>>>(best);
    // 3) WMMA GEMM + fused argmax (35 x 70 tiles of 256N x 128M)
    gemm_argmax<<<dim3(LP / 256, LP / 128), 256, 0, stream>>>(Abf, Bbf, best);
    // 4) gather matched patches + per-patch SSE
    gather_mse<<<LL, 256, 0, stream>>>(x, style, best, partials);
    // 5) deterministic final mean
    finalize<<<1, 256, 0, stream>>>(partials, out);
}
